// GAT_80719615361184
// MI455X (gfx1250) — matrix operand order repaired and checked by reading, not compiled, not run on hardware
//
#include <hip/hip_runtime.h>
#include <hip/hip_bf16.h>

// ---------------- problem constants ----------------
#define N_NODES  50000
#define N_EDGES  800000
#define E_TOT    850000        // + self loops
#define IN_DIM   128
#define HEADS    4
#define CH       64            // hidden per head
#define HD       256           // HEADS*CH
#define OUT_DIM  64
#define EPS_BN   1e-5f
#define NEG_SLOPE 0.2f

typedef __attribute__((ext_vector_type(16))) __bf16 v16bf;
typedef __attribute__((ext_vector_type(8)))  __bf16 v8bf;
typedef __attribute__((ext_vector_type(8)))  float  v8f;

// float -> bf16 round-to-nearest-even (bit trick; NaN not expected here)
__device__ __forceinline__ unsigned short f2bf(float f) {
  unsigned u = __float_as_uint(f);
  u += 0x7FFFu + ((u >> 16) & 1u);
  return (unsigned short)(u >> 16);
}
// order-preserving float<->uint mapping so atomicMax(unsigned) == float max
__device__ __forceinline__ unsigned flip_f(float f) {
  unsigned u = __float_as_uint(f);
  return u ^ ((unsigned)(-(int)(u >> 31)) | 0x80000000u);
}
__device__ __forceinline__ float unflip_f(unsigned u) {
  u ^= ((u >> 31) ? 0x80000000u : 0xFFFFFFFFu);
  return __uint_as_float(u);
}

// ---------------- utility fills ----------------
__global__ void fill_u32_kernel(unsigned* __restrict__ p, unsigned v, long n) {
  long i = (long)blockIdx.x * blockDim.x + threadIdx.x;
  if (i < n) p[i] = v;
}

// W[K][Nn] (row-major, K x Nn) -> WT_bf16[Nn][K]
__global__ void tconv_kernel(const float* __restrict__ src, unsigned short* __restrict__ dst,
                             int K, int Nn) {
  int i = blockIdx.x * blockDim.x + threadIdx.x;
  if (i >= K * Nn) return;
  int n = i / K, k = i - n * K;
  dst[i] = f2bf(src[(size_t)k * Nn + n]);
}

// ---------------- BatchNorm (training-mode, biased var) ----------------
// blockDim.x == Cdim (power of two <= 1024); thread t owns column t -> coalesced
__global__ void bn_partial_kernel(const float* __restrict__ x, const float* __restrict__ bias,
                                  float* __restrict__ gsum, float* __restrict__ gsq, int Cdim) {
  int c = threadIdx.x;
  float b = bias ? bias[c] : 0.f;
  float s = 0.f, q = 0.f;
  for (int r = blockIdx.x; r < N_NODES; r += gridDim.x) {
    float v = x[(size_t)r * Cdim + c] + b;
    s += v; q += v * v;
  }
  atomicAdd(&gsum[c], s);
  atomicAdd(&gsq[c], q);
}

__global__ void bn_finalize_kernel(const float* __restrict__ gsum, const float* __restrict__ gsq,
                                   float* __restrict__ mean, float* __restrict__ rstd, int Cdim) {
  int c = threadIdx.x;
  if (c < Cdim) {
    float m = gsum[c] * (1.f / (float)N_NODES);
    float v = gsq[c] * (1.f / (float)N_NODES) - m * m;
    mean[c] = m;
    rstd[c] = rsqrtf(v + EPS_BN);
  }
}

// normalize (+bias pre-add, +relu) and convert to bf16 for the WMMA A-operand
__global__ void bn_apply_kernel(const float* __restrict__ x, const float* __restrict__ bias,
                                const float* __restrict__ mean, const float* __restrict__ rstd,
                                const float* __restrict__ gamma, const float* __restrict__ beta,
                                unsigned short* __restrict__ ybf, int Cdim, long total, int relu) {
  long i = (long)blockIdx.x * blockDim.x + threadIdx.x;
  if (i >= total) return;
  int c = (int)(i & (Cdim - 1));             // Cdim is 128 or 256
  float v = x[i] + (bias ? bias[c] : 0.f);
  v = (v - mean[c]) * rstd[c] * gamma[c] + beta[c];
  if (relu) v = fmaxf(v, 0.f);
  ybf[i] = f2bf(v);
}

// ---------------- WMMA bf16 GEMM:  C[M x Nn] = A_bf16[M x K] * B  (+bias) ----
// B supplied pre-transposed as BT_bf16[Nn][K] so each lane reads 16 contiguous K.
// One wave per 16x16 output tile; blockDim = (32,4); grid = (M/16, Nn/64).
__global__ void wmma_gemm_kernel(const unsigned short* __restrict__ Abf,
                                 const unsigned short* __restrict__ BTbf,
                                 const float* __restrict__ bias,
                                 float* __restrict__ Cmat, int K, int Nn) {
  const int lane   = threadIdx.x;                       // 0..31
  const int tile_n = blockIdx.y * blockDim.y + threadIdx.y;
  const int tile_m = blockIdx.x;
  const int l15    = lane & 15;
  const int sel    = lane >> 4;                          // which K-half this lane holds

  const __bf16* Arow = (const __bf16*)Abf + (size_t)((tile_m << 4) + l15) * K;
  const __bf16* Brow = (const __bf16*)BTbf + (size_t)((tile_n << 4) + l15) * K + sel * 16;

  v8f acc = {};
  for (int k0 = 0; k0 < K; k0 += 32) {
    // A 16x32 bf16 layout: lane L holds row (L&15); elems 0..15 = K[k0+sel*16 .. +16)
    // contiguous (same k order as B)
    union { v16bf v; struct { v8bf lo; v8bf hi; } s; } a;
    a.s.lo = *(const v8bf*)(Arow + k0 + sel * 16);
    a.s.hi = *(const v8bf*)(Arow + k0 + sel * 16 + 8);
    // B 32x16 bf16 layout: lane L holds col (L&15), K[k0+sel*16 .. +16) contiguous
    v16bf b = *(const v16bf*)(Brow + k0);
    acc = __builtin_amdgcn_wmma_f32_16x16x32_bf16(
        /*neg_a=*/false, a.v, /*neg_b=*/false, b,
        /*c_mod=*/(short)0, acc, /*reuse_a=*/false, /*reuse_b=*/false);
  }
  // C/D f32 layout: VGPR r -> (M = r + sel*8, N = lane&15)
  const int cn = (tile_n << 4) + l15;
  const float bv = bias ? bias[cn] : 0.f;
  float* outp = Cmat + (size_t)((tile_m << 4) + sel * 8) * Nn + cn;
#pragma unroll
  for (int r = 0; r < 8; ++r) outp[(size_t)r * Nn] = acc[r] + bv;
}

// ---------------- attention coefficients: a[n][h] = <hl[n,h,:], att[h,:]> ----
// one wave per (node, head); blockDim = (32,8)
__global__ void att_coeff_kernel(const float* __restrict__ hl,
                                 const float* __restrict__ att_src,
                                 const float* __restrict__ att_dst,
                                 float* __restrict__ a_src, float* __restrict__ a_dst) {
  int wid = blockIdx.x * blockDim.y + threadIdx.y;      // (node*HEADS + head)
  if (wid >= N_NODES * HEADS) return;
  int lane = threadIdx.x;
  int n = wid >> 2, h = wid & 3;
  const float* row = hl + (size_t)n * HD + h * CH;
  const float* as  = att_src + h * CH;
  const float* ad  = att_dst + h * CH;
  float s = row[lane] * as[lane] + row[lane + 32] * as[lane + 32];
  float d = row[lane] * ad[lane] + row[lane + 32] * ad[lane + 32];
#pragma unroll
  for (int off = 16; off > 0; off >>= 1) {
    s += __shfl_xor(s, off, 32);
    d += __shfl_xor(d, off, 32);
  }
  if (lane == 0) { a_src[wid] = s; a_dst[wid] = d; }
}

// ---------------- edge kernels (segment softmax over dst) ----------------
__device__ __forceinline__ void edge_sd(const long long* __restrict__ ei, int e, int& s, int& d) {
  if (e < N_EDGES) { s = (int)ei[e]; d = (int)ei[N_EDGES + e]; }
  else             { s = d = e - N_EDGES; }               // self loop
}

__global__ void edge_max_kernel(const long long* __restrict__ ei,
                                const float* __restrict__ a_src, const float* __restrict__ a_dst,
                                unsigned* __restrict__ nmax) {
  int t = blockIdx.x * blockDim.x + threadIdx.x;
  if (t >= E_TOT * HEADS) return;
  int e = t >> 2, h = t & 3, s, d;
  edge_sd(ei, e, s, d);
  float v = a_src[s * HEADS + h] + a_dst[d * HEADS + h];
  v = v > 0.f ? v : NEG_SLOPE * v;
  atomicMax(&nmax[d * HEADS + h], flip_f(v));
}

__global__ void edge_exp_kernel(const long long* __restrict__ ei,
                                const float* __restrict__ a_src, const float* __restrict__ a_dst,
                                const unsigned* __restrict__ nmax,
                                float* __restrict__ denom, float* __restrict__ expe) {
  int t = blockIdx.x * blockDim.x + threadIdx.x;
  if (t >= E_TOT * HEADS) return;
  int e = t >> 2, h = t & 3, s, d;
  edge_sd(ei, e, s, d);
  float v = a_src[s * HEADS + h] + a_dst[d * HEADS + h];
  v = v > 0.f ? v : NEG_SLOPE * v;
  float w = __expf(v - unflip_f(nmax[d * HEADS + h]));
  expe[t] = w;
  atomicAdd(&denom[d * HEADS + h], w);
}

// one wave per edge: 256 channels, 8 per lane; blockDim = (32,8)
__global__ void edge_agg_kernel(const long long* __restrict__ ei,
                                const float* __restrict__ expe, const float* __restrict__ denom,
                                const float* __restrict__ hl, float* __restrict__ agg) {
  int e = blockIdx.x * blockDim.y + threadIdx.y;
  if (e >= E_TOT) return;
  int lane = threadIdx.x, s, d;
  edge_sd(ei, e, s, d);
  float alpha[HEADS];
#pragma unroll
  for (int h = 0; h < HEADS; ++h)
    alpha[h] = expe[(size_t)e * HEADS + h] / denom[(size_t)d * HEADS + h];
  const float* src = hl + (size_t)s * HD;
  float* dst = agg + (size_t)d * HD;
#pragma unroll
  for (int it = 0; it < 8; ++it) {
    int idx = it * 32 + lane;                       // head = it>>1
    atomicAdd(&dst[idx], alpha[it >> 1] * src[idx]);
  }
}

// ---------------- final: mean over heads + bias2 ----------------
__global__ void final_mean_kernel(const float* __restrict__ agg, const float* __restrict__ bias2,
                                  float* __restrict__ out) {
  long i = (long)blockIdx.x * blockDim.x + threadIdx.x;
  if (i >= (long)N_NODES * OUT_DIM) return;
  int n = (int)(i >> 6), c = (int)(i & 63);
  const float* row = agg + (size_t)n * HD;
  float s = row[c] + row[CH + c] + row[2 * CH + c] + row[3 * CH + c];
  out[i] = 0.25f * s + bias2[c];
}

// ---------------- host orchestration ----------------
static inline size_t al256(size_t x) { return (x + 255) & ~(size_t)255; }
static inline int cdiv(long a, int b) { return (int)((a + b - 1) / b); }

extern "C" void kernel_launch(void* const* d_in, const int* in_sizes, int n_in,
                              void* d_out, int out_size, void* d_ws, size_t ws_size,
                              hipStream_t stream) {
  const float*     x        = (const float*)d_in[0];
  const long long* ei       = (const long long*)d_in[1];   // int64 edge_index [2,E]
  const float*     gamma1   = (const float*)d_in[2];
  const float*     beta1    = (const float*)d_in[3];
  const float*     W1       = (const float*)d_in[4];
  const float*     att_src1 = (const float*)d_in[5];
  const float*     att_dst1 = (const float*)d_in[6];
  const float*     bias1    = (const float*)d_in[7];
  const float*     gamma2   = (const float*)d_in[8];
  const float*     beta2    = (const float*)d_in[9];
  const float*     W2       = (const float*)d_in[10];
  const float*     att_src2 = (const float*)d_in[11];
  const float*     att_dst2 = (const float*)d_in[12];
  const float*     bias2    = (const float*)d_in[13];
  float* out = (float*)d_out;

  // workspace layout
  char* p = (char*)d_ws;
  size_t off = 0;
  auto alloc = [&](size_t bytes) { void* q = p + off; off = al256(off + bytes); return q; };
  unsigned short* w1t  = (unsigned short*)alloc((size_t)HD * IN_DIM * 2);
  unsigned short* w2t  = (unsigned short*)alloc((size_t)HD * HD * 2);
  float* gsum  = (float*)alloc(256 * 4);
  float* gsq   = (float*)alloc(256 * 4);
  float* meanv = (float*)alloc(256 * 4);
  float* rstdv = (float*)alloc(256 * 4);
  float* a_src = (float*)alloc((size_t)N_NODES * HEADS * 4);
  float* a_dst = (float*)alloc((size_t)N_NODES * HEADS * 4);
  unsigned* nmax = (unsigned*)alloc((size_t)N_NODES * HEADS * 4);
  float* denom = (float*)alloc((size_t)N_NODES * HEADS * 4);
  float* expe  = (float*)alloc((size_t)E_TOT * HEADS * 4);
  unsigned short* xbf = (unsigned short*)alloc((size_t)N_NODES * HD * 2);  // bf16 GEMM input (both layers)
  float* hl  = (float*)alloc((size_t)N_NODES * HD * 4);                    // GEMM output (both layers)
  float* agg = (float*)alloc((size_t)N_NODES * HD * 4);                    // aggregation (both layers)
  (void)ws_size; (void)in_sizes; (void)n_in; (void)out_size;

  const long NH   = (long)N_NODES * HEADS;
  const long EH   = (long)E_TOT * HEADS;
  const long NHD  = (long)N_NODES * HD;
  const dim3 gemmGrid(N_NODES / 16, HD / 64), gemmBlock(32, 4);
  const dim3 waveGrid8(32, 8);

  // ---- weight transpose + bf16 convert ----
  tconv_kernel<<<cdiv(IN_DIM * HD, 256), 256, 0, stream>>>(W1, w1t, IN_DIM, HD);
  tconv_kernel<<<cdiv(HD * HD, 256), 256, 0, stream>>>(W2, w2t, HD, HD);

  // ---- BN1 -> bf16 ----
  fill_u32_kernel<<<2, 256, 0, stream>>>((unsigned*)gsum, 0u, 512);  // gsum+gsq contiguous
  bn_partial_kernel<<<256, IN_DIM, 0, stream>>>(x, nullptr, gsum, gsq, IN_DIM);
  bn_finalize_kernel<<<1, IN_DIM, 0, stream>>>(gsum, gsq, meanv, rstdv, IN_DIM);
  bn_apply_kernel<<<cdiv((long)N_NODES * IN_DIM, 256), 256, 0, stream>>>(
      x, nullptr, meanv, rstdv, gamma1, beta1, xbf, IN_DIM, (long)N_NODES * IN_DIM, 0);

  // ---- GAT layer 1 ----
  wmma_gemm_kernel<<<gemmGrid, gemmBlock, 0, stream>>>(xbf, w1t, bias1, hl, IN_DIM, HD);
  att_coeff_kernel<<<cdiv(NH, 8), waveGrid8, 0, stream>>>(hl, att_src1, att_dst1, a_src, a_dst);
  fill_u32_kernel<<<cdiv(NH, 256), 256, 0, stream>>>(nmax, 0x007FFFFFu, NH);   // flip(-inf)
  fill_u32_kernel<<<cdiv(NH, 256), 256, 0, stream>>>((unsigned*)denom, 0u, NH);
  fill_u32_kernel<<<cdiv(NHD, 256), 256, 0, stream>>>((unsigned*)agg, 0u, NHD);
  edge_max_kernel<<<cdiv(EH, 256), 256, 0, stream>>>(ei, a_src, a_dst, nmax);
  edge_exp_kernel<<<cdiv(EH, 256), 256, 0, stream>>>(ei, a_src, a_dst, nmax, denom, expe);
  edge_agg_kernel<<<cdiv(E_TOT, 8), waveGrid8, 0, stream>>>(ei, expe, denom, hl, agg);

  // ---- BN2 (+bias1 pre-add) + relu -> bf16 ----
  fill_u32_kernel<<<2, 256, 0, stream>>>((unsigned*)gsum, 0u, 512);
  bn_partial_kernel<<<256, HD, 0, stream>>>(agg, bias1, gsum, gsq, HD);
  bn_finalize_kernel<<<1, HD, 0, stream>>>(gsum, gsq, meanv, rstdv, HD);
  bn_apply_kernel<<<cdiv(NHD, 256), 256, 0, stream>>>(
      agg, bias1, meanv, rstdv, gamma2, beta2, xbf, HD, NHD, 1);

  // ---- GAT layer 2 (reuse hl / agg) ----
  wmma_gemm_kernel<<<gemmGrid, gemmBlock, 0, stream>>>(xbf, w2t, nullptr, hl, HD, HD);
  att_coeff_kernel<<<cdiv(NH, 8), waveGrid8, 0, stream>>>(hl, att_src2, att_dst2, a_src, a_dst);
  fill_u32_kernel<<<cdiv(NH, 256), 256, 0, stream>>>(nmax, 0x007FFFFFu, NH);
  fill_u32_kernel<<<cdiv(NH, 256), 256, 0, stream>>>((unsigned*)denom, 0u, NH);
  fill_u32_kernel<<<cdiv(NHD, 256), 256, 0, stream>>>((unsigned*)agg, 0u, NHD);
  edge_max_kernel<<<cdiv(EH, 256), 256, 0, stream>>>(ei, a_src, a_dst, nmax);
  edge_exp_kernel<<<cdiv(EH, 256), 256, 0, stream>>>(ei, a_src, a_dst, nmax, denom, expe);
  edge_agg_kernel<<<cdiv(E_TOT, 8), waveGrid8, 0, stream>>>(ei, expe, denom, hl, agg);

  // ---- head mean + bias2 ----
  final_mean_kernel<<<cdiv((long)N_NODES * OUT_DIM, 256), 256, 0, stream>>>(agg, bias2, out);
}
